// Transformer_Block_12584254177955
// MI455X (gfx1250) — matrix operand order repaired and checked by reading, not compiled, not run on hardware
//
#include <hip/hip_runtime.h>

// ---------------------------------------------------------------------------
// Transformer block for MI455X (gfx1250, wave32, WMMA).
// All matmuls use v_wmma_f32_16x16x32_bf16 (fp32 accumulate).
// GEMMs: 64x64 wave tiles; the 64x32 A k-panel is staged once per block into
// LDS via GLOBAL_LOAD_ASYNC_TO_LDS_B128 (ASYNCcnt), double-buffered.
// ---------------------------------------------------------------------------

#define D_MODEL   1024
#define D_QKV     3072
#define NUM_HEADS 16
#define HEAD_DIM  64
#define D_FF      2688
#define BATCH     2
#define SEQ       2048
#define TOKENS    (BATCH * SEQ)
#define EPS       1e-5f

#define HAS_ASYNC_LDS __has_builtin(__builtin_amdgcn_global_load_async_to_lds_b128)
#define HAS_WAIT_ASYNC __has_builtin(__builtin_amdgcn_s_wait_asynccnt)

typedef __bf16 bf16;
typedef __attribute__((ext_vector_type(8)))  __bf16         v8bf;
typedef __attribute__((ext_vector_type(16))) __bf16         v16bf;
typedef __attribute__((ext_vector_type(8)))  float          v8f;
typedef __attribute__((ext_vector_type(8)))  unsigned short v8u;
typedef __attribute__((ext_vector_type(4)))  int            i4;

// ---- fp32 -> bf16 bit conversion (round-to-nearest-even) -------------------
static __device__ __forceinline__ unsigned short f2bf(float f) {
  unsigned int u = __builtin_bit_cast(unsigned int, f);
  unsigned int r = (u + 0x7FFFu + ((u >> 16) & 1u)) >> 16;
  return (unsigned short)r;
}

static __device__ __forceinline__ v8f wmma_bf16(v16bf a, v16bf b, v8f c) {
  return __builtin_amdgcn_wmma_f32_16x16x32_bf16(false, a, false, b,
                                                 (short)0, c, false, false);
}

// ---------------------------------------------------------------------------
// Fragment loaders: A and B both use contiguous k order (k = 16*(lane/16) + s).
// ---------------------------------------------------------------------------
static __device__ __forceinline__ v16bf load_a_frag(const bf16* base, int ld, int k0) {
  const int lane = threadIdx.x & 31;
  const int m  = lane & 15;
  const int hi = lane >> 4;
  const bf16* p = base + (size_t)m * ld + k0 + hi * 16;
  v8bf x0 = *(const v8bf*)p;
  v8bf x1 = *(const v8bf*)(p + 8);
  v16bf r;
#pragma unroll
  for (int i = 0; i < 8; ++i) { r[i] = x0[i]; r[i + 8] = x1[i]; }
  return r;
}

static __device__ __forceinline__ v16bf load_b_frag(const bf16* base, int ld, int k0) {
  const int lane = threadIdx.x & 31;
  const int n    = lane & 15;
  const int koff = (lane >> 4) * 16;
  return *(const v16bf*)(base + (size_t)n * ld + k0 + koff);
}

// ---------------------------------------------------------------------------
// Async staging helpers.  Builtin signature (confirmed by round-2 diagnostic):
//   __builtin_amdgcn_global_load_async_to_lds_b128(int4 AS1* src,
//                                                  int4 AS3* dst,
//                                                  imm offset, imm cpol)
// ---------------------------------------------------------------------------
#if HAS_ASYNC_LDS
static __device__ __forceinline__ void async_copy_b128(const void* g, void* l) {
  __builtin_amdgcn_global_load_async_to_lds_b128(
      (__attribute__((address_space(1))) i4*)g,
      (__attribute__((address_space(3))) i4*)l, 0, 0);
}
static __device__ __forceinline__ void wait_async0() {
#if HAS_WAIT_ASYNC
  __builtin_amdgcn_s_wait_asynccnt(0);
#else
  asm volatile("s_wait_asynccnt 0x0" ::: "memory");
#endif
}
#endif

// ---------------------------------------------------------------------------
// fp32 -> bf16 elementwise (weights)
// ---------------------------------------------------------------------------
__global__ __launch_bounds__(256) void cvt_kernel(const float* __restrict__ s,
                                                  bf16* __restrict__ d, int n4) {
  const int i = blockIdx.x * 256 + threadIdx.x;
  if (i >= n4) return;
  const float4 v = ((const float4*)s)[i];
  unsigned short* o = (unsigned short*)d + (size_t)i * 4;
  o[0] = f2bf(v.x); o[1] = f2bf(v.y); o[2] = f2bf(v.z); o[3] = f2bf(v.w);
}

// ---------------------------------------------------------------------------
// RMSNorm: one row (1024) per 256-thread block; bf16 output.
// ---------------------------------------------------------------------------
__global__ __launch_bounds__(256) void rmsnorm_kernel(const float* __restrict__ x,
                                                      const float* __restrict__ g,
                                                      bf16* __restrict__ out) {
  __shared__ float red[8];
  const int row = blockIdx.x;
  const int tid = threadIdx.x;
  const float4 v = ((const float4*)(x + (size_t)row * D_MODEL))[tid];
  float ss = v.x * v.x + v.y * v.y + v.z * v.z + v.w * v.w;
#pragma unroll
  for (int m = 1; m < 32; m <<= 1) ss += __shfl_xor(ss, m, 32);
  if ((tid & 31) == 0) red[tid >> 5] = ss;
  __syncthreads();
  float tot = 0.f;
#pragma unroll
  for (int i = 0; i < 8; ++i) tot += red[i];
  const float rinv = rsqrtf(tot * (1.0f / D_MODEL) + EPS);
  const float4 gv = ((const float4*)g)[tid];
  unsigned short* o = (unsigned short*)out + (size_t)row * D_MODEL + tid * 4;
  o[0] = f2bf(v.x * gv.x * rinv);
  o[1] = f2bf(v.y * gv.y * rinv);
  o[2] = f2bf(v.z * gv.z * rinv);
  o[3] = f2bf(v.w * gv.w * rinv);
}

// ---------------------------------------------------------------------------
// Block-cooperative WMMA GEMM:  C[M,N] = A[M,K](bf16) x W[N,K]^T(bf16).
// Block = 8 waves, block tile 64x512, wave tile 64x64 (16 accumulators).
// A k-panel (64x32) staged once per block into LDS, double-buffered, via
// async-to-LDS; weights stream from global (all fit in the 192MB L2).
// Requires N % 512 == 0.
// mode 0: store bf16.  mode 1: fp32 out = acc + res.  mode 2: fp32 out += acc.
// ---------------------------------------------------------------------------
#define APAD 40  // padded row stride of the LDS A panel (bank spread)

__global__ __launch_bounds__(256) void gemm_kernel(const bf16* __restrict__ A,
                                                   const bf16* __restrict__ W,
                                                   void* __restrict__ out,
                                                   const float* __restrict__ res,
                                                   int N, int K, int mode) {
  __shared__ __align__(128) bf16 apan[2][64 * APAD];
  const int wave = threadIdx.x >> 5;
  const int lane = threadIdx.x & 31;
  const int ln = lane & 15;
  const int hi = lane >> 4;

  const int nb_n = N >> 9;                  // N / 512
  const int bm = blockIdx.x / nb_n;
  const int bn = blockIdx.x - bm * nb_n;
  const int m0 = bm * 64;
  const int n0 = bn * 512 + wave * 64;

  const bf16* Ab = A + (size_t)m0 * K;
  const bf16* Wb = W + (size_t)n0 * K;

  // staging assignment: each of the 256 threads moves one 16B chunk per panel
  const int ar = threadIdx.x >> 2;          // row 0..63
  const int ac = (threadIdx.x & 3) * 8;     // col 0,8,16,24

  v8f c[4][4];
#pragma unroll
  for (int i = 0; i < 4; ++i)
#pragma unroll
    for (int j = 0; j < 4; ++j) c[i][j] = (v8f){};

  const int nk = K >> 5;

#if HAS_ASYNC_LDS
  async_copy_b128(Ab + (size_t)ar * K + ac, &apan[0][ar * APAD + ac]);
  wait_async0();
  __syncthreads();
  for (int kt = 0; kt < nk; ++kt) {
    const bf16* cur = apan[kt & 1];
    if (kt + 1 < nk)
      async_copy_b128(Ab + (size_t)ar * K + (kt + 1) * 32 + ac,
                      &apan[(kt + 1) & 1][ar * APAD + ac]);
    const int k = kt * 32;
    v16bf a0 = load_a_frag(cur,             APAD, 0);
    v16bf a1 = load_a_frag(cur + 16 * APAD, APAD, 0);
    v16bf a2 = load_a_frag(cur + 32 * APAD, APAD, 0);
    v16bf a3 = load_a_frag(cur + 48 * APAD, APAD, 0);
    v16bf b0 = load_b_frag(Wb,                  K, k);
    v16bf b1 = load_b_frag(Wb + (size_t)16 * K, K, k);
    v16bf b2 = load_b_frag(Wb + (size_t)32 * K, K, k);
    v16bf b3 = load_b_frag(Wb + (size_t)48 * K, K, k);
    c[0][0] = wmma_bf16(a0, b0, c[0][0]); c[0][1] = wmma_bf16(a0, b1, c[0][1]);
    c[0][2] = wmma_bf16(a0, b2, c[0][2]); c[0][3] = wmma_bf16(a0, b3, c[0][3]);
    c[1][0] = wmma_bf16(a1, b0, c[1][0]); c[1][1] = wmma_bf16(a1, b1, c[1][1]);
    c[1][2] = wmma_bf16(a1, b2, c[1][2]); c[1][3] = wmma_bf16(a1, b3, c[1][3]);
    c[2][0] = wmma_bf16(a2, b0, c[2][0]); c[2][1] = wmma_bf16(a2, b1, c[2][1]);
    c[2][2] = wmma_bf16(a2, b2, c[2][2]); c[2][3] = wmma_bf16(a2, b3, c[2][3]);
    c[3][0] = wmma_bf16(a3, b0, c[3][0]); c[3][1] = wmma_bf16(a3, b1, c[3][1]);
    c[3][2] = wmma_bf16(a3, b2, c[3][2]); c[3][3] = wmma_bf16(a3, b3, c[3][3]);
    wait_async0();
    __syncthreads();
  }
#else
  {
    uint4 r = *(const uint4*)(Ab + (size_t)ar * K + ac);
    *(uint4*)&apan[0][ar * APAD + ac] = r;
  }
  __syncthreads();
  for (int kt = 0; kt < nk; ++kt) {
    const bf16* cur = apan[kt & 1];
    bf16* nxt = apan[(kt + 1) & 1];
    uint4 r;
    if (kt + 1 < nk)
      r = *(const uint4*)(Ab + (size_t)ar * K + (kt + 1) * 32 + ac);
    const int k = kt * 32;
    v16bf a0 = load_a_frag(cur,             APAD, 0);
    v16bf a1 = load_a_frag(cur + 16 * APAD, APAD, 0);
    v16bf a2 = load_a_frag(cur + 32 * APAD, APAD, 0);
    v16bf a3 = load_a_frag(cur + 48 * APAD, APAD, 0);
    v16bf b0 = load_b_frag(Wb,                  K, k);
    v16bf b1 = load_b_frag(Wb + (size_t)16 * K, K, k);
    v16bf b2 = load_b_frag(Wb + (size_t)32 * K, K, k);
    v16bf b3 = load_b_frag(Wb + (size_t)48 * K, K, k);
    c[0][0] = wmma_bf16(a0, b0, c[0][0]); c[0][1] = wmma_bf16(a0, b1, c[0][1]);
    c[0][2] = wmma_bf16(a0, b2, c[0][2]); c[0][3] = wmma_bf16(a0, b3, c[0][3]);
    c[1][0] = wmma_bf16(a1, b0, c[1][0]); c[1][1] = wmma_bf16(a1, b1, c[1][1]);
    c[1][2] = wmma_bf16(a1, b2, c[1][2]); c[1][3] = wmma_bf16(a1, b3, c[1][3]);
    c[2][0] = wmma_bf16(a2, b0, c[2][0]); c[2][1] = wmma_bf16(a2, b1, c[2][1]);
    c[2][2] = wmma_bf16(a2, b2, c[2][2]); c[2][3] = wmma_bf16(a2, b3, c[2][3]);
    c[3][0] = wmma_bf16(a3, b0, c[3][0]); c[3][1] = wmma_bf16(a3, b1, c[3][1]);
    c[3][2] = wmma_bf16(a3, b2, c[3][2]); c[3][3] = wmma_bf16(a3, b3, c[3][3]);
    __syncthreads();
    if (kt + 1 < nk) *(uint4*)&nxt[ar * APAD + ac] = r;
    __syncthreads();
  }
#endif

  if (mode == 0) {
    unsigned short* ob = (unsigned short*)out;
#pragma unroll
    for (int i = 0; i < 4; ++i)
#pragma unroll
      for (int j = 0; j < 4; ++j)
#pragma unroll
        for (int v = 0; v < 8; ++v) {
          const size_t idx = (size_t)(m0 + i * 16 + v + 8 * hi) * N + n0 + j * 16 + ln;
          ob[idx] = f2bf(c[i][j][v]);
        }
  } else if (mode == 1) {
    float* of = (float*)out;
#pragma unroll
    for (int i = 0; i < 4; ++i)
#pragma unroll
      for (int j = 0; j < 4; ++j)
#pragma unroll
        for (int v = 0; v < 8; ++v) {
          const size_t idx = (size_t)(m0 + i * 16 + v + 8 * hi) * N + n0 + j * 16 + ln;
          of[idx] = c[i][j][v] + res[idx];
        }
  } else {
    float* of = (float*)out;
#pragma unroll
    for (int i = 0; i < 4; ++i)
#pragma unroll
      for (int j = 0; j < 4; ++j)
#pragma unroll
        for (int v = 0; v < 8; ++v) {
          const size_t idx = (size_t)(m0 + i * 16 + v + 8 * hi) * N + n0 + j * 16 + ln;
          of[idx] += c[i][j][v];
        }
  }
}

// ---------------------------------------------------------------------------
// Fused FFN up-projection: a = X@W1^T, b = X@W3^T, H = silu(a)*b (bf16 out).
// Wave tile 64x32 against BOTH weight matrices (16 accumulators).
// ---------------------------------------------------------------------------
__global__ __launch_bounds__(256) void ffn13_kernel(const bf16* __restrict__ X,
                                                    const bf16* __restrict__ W1,
                                                    const bf16* __restrict__ W3,
                                                    bf16* __restrict__ H) {
  const int wave = threadIdx.x >> 5;
  const int lane = threadIdx.x & 31;
  const int ln = lane & 15;
  const int hi = lane >> 4;
  const int tiles_n = D_FF >> 5;           // 84
  const int tile = blockIdx.x * 8 + wave;
  const int mt = tile / tiles_n;           // 64-row tile
  const int nt = tile - mt * tiles_n;      // 32-col tile

  const bf16* Ab  = X  + (size_t)(mt * 64) * D_MODEL;
  const bf16* W1b = W1 + (size_t)(nt * 32) * D_MODEL;
  const bf16* W3b = W3 + (size_t)(nt * 32) * D_MODEL;

  v8f ca[4][2], cb[4][2];
#pragma unroll
  for (int i = 0; i < 4; ++i)
#pragma unroll
    for (int j = 0; j < 2; ++j) { ca[i][j] = (v8f){}; cb[i][j] = (v8f){}; }

  for (int k = 0; k < D_MODEL; k += 32) {
    v16bf a0 = load_a_frag(Ab,                          D_MODEL, k);
    v16bf a1 = load_a_frag(Ab + (size_t)16 * D_MODEL,   D_MODEL, k);
    v16bf a2 = load_a_frag(Ab + (size_t)32 * D_MODEL,   D_MODEL, k);
    v16bf a3 = load_a_frag(Ab + (size_t)48 * D_MODEL,   D_MODEL, k);
    v16bf u0 = load_b_frag(W1b,                         D_MODEL, k);
    v16bf u1 = load_b_frag(W1b + (size_t)16 * D_MODEL,  D_MODEL, k);
    v16bf t0 = load_b_frag(W3b,                         D_MODEL, k);
    v16bf t1 = load_b_frag(W3b + (size_t)16 * D_MODEL,  D_MODEL, k);
    ca[0][0] = wmma_bf16(a0, u0, ca[0][0]); ca[0][1] = wmma_bf16(a0, u1, ca[0][1]);
    ca[1][0] = wmma_bf16(a1, u0, ca[1][0]); ca[1][1] = wmma_bf16(a1, u1, ca[1][1]);
    ca[2][0] = wmma_bf16(a2, u0, ca[2][0]); ca[2][1] = wmma_bf16(a2, u1, ca[2][1]);
    ca[3][0] = wmma_bf16(a3, u0, ca[3][0]); ca[3][1] = wmma_bf16(a3, u1, ca[3][1]);
    cb[0][0] = wmma_bf16(a0, t0, cb[0][0]); cb[0][1] = wmma_bf16(a0, t1, cb[0][1]);
    cb[1][0] = wmma_bf16(a1, t0, cb[1][0]); cb[1][1] = wmma_bf16(a1, t1, cb[1][1]);
    cb[2][0] = wmma_bf16(a2, t0, cb[2][0]); cb[2][1] = wmma_bf16(a2, t1, cb[2][1]);
    cb[3][0] = wmma_bf16(a3, t0, cb[3][0]); cb[3][1] = wmma_bf16(a3, t1, cb[3][1]);
  }

  unsigned short* ob = (unsigned short*)H;
#pragma unroll
  for (int i = 0; i < 4; ++i)
#pragma unroll
    for (int j = 0; j < 2; ++j)
#pragma unroll
      for (int v = 0; v < 8; ++v) {
        const size_t idx =
            (size_t)(mt * 64 + i * 16 + v + 8 * hi) * D_FF + nt * 32 + j * 16 + ln;
        const float a = ca[i][j][v];
        const float hval = a * (1.0f / (1.0f + __expf(-a))) * cb[i][j][v];
        ob[idx] = f2bf(hval);
      }
}

// ---------------------------------------------------------------------------
// Causal flash attention (per-wave LDS, no block sync needed).
// ---------------------------------------------------------------------------
__global__ __launch_bounds__(256) void flash_attn_kernel(const bf16* __restrict__ qkv,
                                                         bf16* __restrict__ out) {
  __shared__ __align__(64) bf16 lds[8 * 2560];  // per wave: P[16][32] + Vt[64][32]
  const int tid  = threadIdx.x;
  const int lane = tid & 31;
  const int wave = tid >> 5;
  const int ln = lane & 15;
  const int hi = lane >> 4;

  const int h     = blockIdx.y;
  const int qb0   = blockIdx.x * 128;
  const int batch = qb0 / SEQ;
  const int sblk  = qb0 % SEQ;
  const int qt0   = qb0 + wave * 16;
  const int sq0   = sblk + wave * 16;

  bf16* pP  = lds + wave * 2560;
  bf16* pVt = pP + 512;
  unsigned int*   vtw = (unsigned int*)pVt;
  unsigned short* pw  = (unsigned short*)pP;

  const int ldq = D_QKV;
  const bf16* qbase = qkv + (size_t)qt0 * ldq + h * HEAD_DIM;
  const v16bf qa0 = load_a_frag(qbase, ldq, 0);
  const v16bf qa1 = load_a_frag(qbase, ldq, 32);

  v8f o0 = {}, o1 = {}, o2 = {}, o3 = {};
  float mrun[8], lrun[8];
#pragma unroll
  for (int v = 0; v < 8; ++v) { mrun[v] = -1e30f; lrun[v] = 0.f; }

  const unsigned short* vsrc = (const unsigned short*)qkv;
  const int ntiles = (sq0 + 16 + 31) / 32;

  for (int kt = 0; kt < ntiles; ++kt) {
    const int ks0 = kt * 32;
    const int kt0 = batch * SEQ + ks0;

    {  // stage V^T into per-wave LDS
      const int j0 = ln * 2;
      const int dh = hi * 32;
      const unsigned short* p0 =
          vsrc + (size_t)(kt0 + j0) * ldq + 2 * D_MODEL + h * HEAD_DIM + dh;
      const unsigned short* p1 = p0 + ldq;
#pragma unroll
      for (int c = 0; c < 4; ++c) {
        v8u r0 = *(const v8u*)(p0 + c * 8);
        v8u r1 = *(const v8u*)(p1 + c * 8);
#pragma unroll
        for (int e = 0; e < 8; ++e) {
          const int d = dh + c * 8 + e;
          vtw[d * 16 + ln] = (unsigned int)r0[e] | ((unsigned int)r1[e] << 16);
        }
      }
    }

    const bf16* kb0 = qkv + (size_t)kt0 * ldq + D_MODEL + h * HEAD_DIM;
    const bf16* kb1 = kb0 + (size_t)16 * ldq;
    v8f s0 = {}, s1 = {};
    s0 = wmma_bf16(qa0, load_b_frag(kb0, ldq, 0), s0);
    s0 = wmma_bf16(qa1, load_b_frag(kb0, ldq, 32), s0);
    s1 = wmma_bf16(qa0, load_b_frag(kb1, ldq, 0), s1);
    s1 = wmma_bf16(qa1, load_b_frag(kb1, ldq, 32), s1);

#pragma unroll
    for (int v = 0; v < 8; ++v) {
      const int qs = sq0 + v + 8 * hi;
      float f0 = s0[v] * 0.125f;
      float f1 = s1[v] * 0.125f;
      if (ks0 + ln > qs)      f0 = -1e30f;
      if (ks0 + 16 + ln > qs) f1 = -1e30f;
      float rm = fmaxf(f0, f1);
      rm = fmaxf(rm, __shfl_xor(rm, 1, 32));
      rm = fmaxf(rm, __shfl_xor(rm, 2, 32));
      rm = fmaxf(rm, __shfl_xor(rm, 4, 32));
      rm = fmaxf(rm, __shfl_xor(rm, 8, 32));
      const float mnew = fmaxf(mrun[v], rm);
      const float corr = __expf(mrun[v] - mnew);
      const float p0 = __expf(f0 - mnew);
      const float p1 = __expf(f1 - mnew);
      float rs = p0 + p1;
      rs += __shfl_xor(rs, 1, 32);
      rs += __shfl_xor(rs, 2, 32);
      rs += __shfl_xor(rs, 4, 32);
      rs += __shfl_xor(rs, 8, 32);
      lrun[v] = lrun[v] * corr + rs;
      mrun[v] = mnew;
      o0[v] *= corr; o1[v] *= corr; o2[v] *= corr; o3[v] *= corr;
      const int prow = (v + 8 * hi) * 32;
      pw[prow + ln]      = f2bf(p0);
      pw[prow + 16 + ln] = f2bf(p1);
    }
    __builtin_amdgcn_wave_barrier();

    const v16bf pa = load_a_frag(pP, 32, 0);
    o0 = wmma_bf16(pa, load_b_frag(pVt,           32, 0), o0);
    o1 = wmma_bf16(pa, load_b_frag(pVt + 16 * 32, 32, 0), o1);
    o2 = wmma_bf16(pa, load_b_frag(pVt + 32 * 32, 32, 0), o2);
    o3 = wmma_bf16(pa, load_b_frag(pVt + 48 * 32, 32, 0), o3);
    __builtin_amdgcn_wave_barrier();
  }

  unsigned short* ob = (unsigned short*)out;
#pragma unroll
  for (int v = 0; v < 8; ++v) {
    const float inv = 1.0f / lrun[v];
    const size_t r = (size_t)(qt0 + v + 8 * hi) * D_MODEL + h * HEAD_DIM + ln;
    ob[r]      = f2bf(o0[v] * inv);
    ob[r + 16] = f2bf(o1[v] * inv);
    ob[r + 32] = f2bf(o2[v] * inv);
    ob[r + 48] = f2bf(o3[v] * inv);
  }
}

// ---------------------------------------------------------------------------
// Host side
// ---------------------------------------------------------------------------
extern "C" void kernel_launch(void* const* d_in, const int* in_sizes, int n_in,
                              void* d_out, int out_size, void* d_ws, size_t ws_size,
                              hipStream_t stream) {
  (void)in_sizes; (void)n_in; (void)out_size; (void)ws_size;
  const float* x    = (const float*)d_in[0];
  const float* wqkv = (const float*)d_in[1];
  const float* wo   = (const float*)d_in[2];
  const float* g1   = (const float*)d_in[3];
  const float* g2   = (const float*)d_in[4];
  const float* w1   = (const float*)d_in[5];
  const float* w2   = (const float*)d_in[6];
  const float* w3   = (const float*)d_in[7];
  float* out = (float*)d_out;

  char* ws = (char*)d_ws;
  size_t off = 0;
  auto take = [&](size_t bytes) -> char* {
    char* p = ws + off;
    off += (bytes + 255) & ~(size_t)255;
    return p;
  };
  bf16* wqkv_b = (bf16*)take((size_t)D_QKV * D_MODEL * 2);
  bf16* wo_b   = (bf16*)take((size_t)D_MODEL * D_MODEL * 2);
  bf16* w1_b   = (bf16*)take((size_t)D_FF * D_MODEL * 2);
  bf16* w3_b   = (bf16*)take((size_t)D_FF * D_MODEL * 2);
  bf16* w2_b   = (bf16*)take((size_t)D_MODEL * D_FF * 2);
  bf16* xn_b   = (bf16*)take((size_t)TOKENS * D_MODEL * 2);
  bf16* qkv_b  = (bf16*)take((size_t)TOKENS * D_QKV * 2);
  bf16* attn_b = (bf16*)take((size_t)TOKENS * D_MODEL * 2);
  bf16* h_b    = (bf16*)take((size_t)TOKENS * D_FF * 2);

  cvt_kernel<<<(D_QKV * D_MODEL / 4 + 255) / 256, 256, 0, stream>>>(wqkv, wqkv_b, D_QKV * D_MODEL / 4);
  cvt_kernel<<<(D_MODEL * D_MODEL / 4 + 255) / 256, 256, 0, stream>>>(wo, wo_b, D_MODEL * D_MODEL / 4);
  cvt_kernel<<<(D_FF * D_MODEL / 4 + 255) / 256, 256, 0, stream>>>(w1, w1_b, D_FF * D_MODEL / 4);
  cvt_kernel<<<(D_FF * D_MODEL / 4 + 255) / 256, 256, 0, stream>>>(w3, w3_b, D_FF * D_MODEL / 4);
  cvt_kernel<<<(D_MODEL * D_FF / 4 + 255) / 256, 256, 0, stream>>>(w2, w2_b, D_MODEL * D_FF / 4);

  rmsnorm_kernel<<<TOKENS, 256, 0, stream>>>(x, g1, xn_b);

  gemm_kernel<<<(TOKENS / 64) * (D_QKV / 512), 256, 0, stream>>>(
      xn_b, wqkv_b, qkv_b, nullptr, D_QKV, D_MODEL, 0);

  flash_attn_kernel<<<dim3(TOKENS / 128, NUM_HEADS), 256, 0, stream>>>(qkv_b, attn_b);

  gemm_kernel<<<(TOKENS / 64) * (D_MODEL / 512), 256, 0, stream>>>(
      attn_b, wo_b, out, x, D_MODEL, D_MODEL, 1);

  rmsnorm_kernel<<<TOKENS, 256, 0, stream>>>(out, g2, xn_b);

  ffn13_kernel<<<(TOKENS / 64) * (D_FF / 32) / 8, 256, 0, stream>>>(xn_b, w1_b, w3_b, h_b);

  gemm_kernel<<<(TOKENS / 64) * (D_MODEL / 512), 256, 0, stream>>>(
      h_b, w2_b, out, nullptr, D_MODEL, D_FF, 2);
}
